// MultiHeadAttentionLayer_13924283974442
// MI455X (gfx1250) — compile-verified, matrix operand order checked by reading, not run on hardware
//
#include <hip/hip_runtime.h>
#include <hip/hip_bf16.h>
#include <stdint.h>

// ---------------------------------------------------------------------------
// MI455X / gfx1250 multi-head attention forward.
// All matmuls run on v_wmma_f32_16x16x32_bf16 (f32 accumulate).
// K-chunk staging uses GLOBAL_LOAD_ASYNC_TO_LDS_B128 when the builtin exists.
// ---------------------------------------------------------------------------

typedef __attribute__((ext_vector_type(16))) __bf16 v16bf;
typedef __attribute__((ext_vector_type(8)))  __bf16 bf16x8;
typedef __attribute__((ext_vector_type(8)))  float  v8f;
typedef __attribute__((ext_vector_type(4)))  int    v4i;

#define EMBED 1024
#define SEQ   2048
#define NHEAD 16
#define DHEAD 64
#define BATCH 2

#if defined(__has_builtin)
#if __has_builtin(__builtin_amdgcn_global_load_async_to_lds_b128)
#define HAVE_ASYNC_LDS 1
#endif
#endif

#define GAS __attribute__((address_space(1)))
#define LAS __attribute__((address_space(3)))

// Native hardware convert (v_cvt_*bf16_f32 class) instead of manual RNE math.
__device__ __forceinline__ __bf16 f2bf(float f) { return (__bf16)f; }

__device__ __forceinline__ v8f zero8() {
  v8f z = {0.f, 0.f, 0.f, 0.f, 0.f, 0.f, 0.f, 0.f};
  return z;
}

// Build a 16-element bf16 A/B fragment for v_wmma_f32_16x16x32_bf16.
// Per ISA layout: lane-half 0 holds K {0..7, 16..23}, lane-half 1 holds
// K {8..15, 24..31}; two aligned 16B loads from a 32-element K row.
__device__ __forceinline__ v16bf load_frag(const __bf16* row, int half) {
  union { v16bf v; bf16x8 h[2]; } u;
  u.h[0] = *(const bf16x8*)(row + half * 8);
  u.h[1] = *(const bf16x8*)(row + 16 + half * 8);
  return u.v;
}

__device__ __forceinline__ v8f wmma_bf16(v16bf a, v16bf b, v8f c) {
  return __builtin_amdgcn_wmma_f32_16x16x32_bf16(false, a, false, b,
                                                 (short)0, c, false, false);
}

__device__ __forceinline__ void store_out(float* p, float v)  { *p = v; }
__device__ __forceinline__ void store_out(__bf16* p, float v) { *p = f2bf(v); }

// ---------------------------------------------------------------------------
// Weight convert + transpose: Wt[n][k] = bf16(W[k][n]).  32x32 LDS tiles.
// ---------------------------------------------------------------------------
__global__ __launch_bounds__(256)
void wt_convert_kernel(const float* __restrict__ W, __bf16* __restrict__ Wt) {
  __shared__ float tile[32][33];
  int n0 = blockIdx.x * 32, k0 = blockIdx.y * 32;
  int tx = threadIdx.x, ty = threadIdx.y;          // 32 x 8
#pragma unroll
  for (int i = ty; i < 32; i += 8)
    tile[i][tx] = W[(size_t)(k0 + i) * EMBED + n0 + tx];
  __syncthreads();
#pragma unroll
  for (int i = ty; i < 32; i += 8)
    Wt[(size_t)(n0 + i) * EMBED + k0 + tx] = f2bf(tile[tx][i]);
}

// ---------------------------------------------------------------------------
// GEMM + bias: C[M,N] = A[M,K](f32, converted to bf16 in staging) @ Wt^T + b.
// Bt is pre-transposed bf16 [N][K].  Block tile 128x128, 8 waves (4x2),
// wave tile 32x64 -> 2x4 16x16 accumulators, K-step 32.
// ---------------------------------------------------------------------------
template <typename OutT>
__global__ __launch_bounds__(256)
void gemm_bias_kernel(const float* __restrict__ A, const __bf16* __restrict__ Bt,
                      const float* __restrict__ bias, OutT* __restrict__ C,
                      int M, int N, int K) {
  __shared__ __bf16 la[128][32];   // A tile, row stride 64B
  __shared__ __bf16 lb[128][32];   // B tile (rows = N columns), row stride 64B

  const int m0 = blockIdx.y * 128, n0 = blockIdx.x * 128;
  const int tid = threadIdx.x;
  const int wave = tid >> 5, lane = tid & 31;
  const int half = lane >> 4, l16 = lane & 15;
  const int wm = wave >> 1, wn = wave & 1;

  v8f acc[2][4];
#pragma unroll
  for (int i = 0; i < 2; ++i)
#pragma unroll
    for (int j = 0; j < 4; ++j) acc[i][j] = zero8();

  const int srow = tid >> 1, shalf = tid & 1;       // 128 rows x 2 half-rows
  const float*  arow = A  + (size_t)(m0 + srow) * K + shalf * 16;
  const __bf16* brow = Bt + (size_t)(n0 + srow) * K + shalf * 16;

  for (int k0 = 0; k0 < K; k0 += 32) {
    // ---- stage A (f32 -> bf16, hardware convert) ----
    const float4* ap = (const float4*)(arow + k0);
    float4 x0 = ap[0], x1 = ap[1], x2 = ap[2], x3 = ap[3];
    bf16x8 alo, ahi;
    alo[0] = f2bf(x0.x); alo[1] = f2bf(x0.y); alo[2] = f2bf(x0.z); alo[3] = f2bf(x0.w);
    alo[4] = f2bf(x1.x); alo[5] = f2bf(x1.y); alo[6] = f2bf(x1.z); alo[7] = f2bf(x1.w);
    ahi[0] = f2bf(x2.x); ahi[1] = f2bf(x2.y); ahi[2] = f2bf(x2.z); ahi[3] = f2bf(x2.w);
    ahi[4] = f2bf(x3.x); ahi[5] = f2bf(x3.y); ahi[6] = f2bf(x3.z); ahi[7] = f2bf(x3.w);
    *(bf16x8*)&la[srow][shalf * 16]     = alo;
    *(bf16x8*)&la[srow][shalf * 16 + 8] = ahi;
    // ---- stage B (already bf16, row-major in K) ----
    const bf16x8* bp = (const bf16x8*)(brow + k0);
    *(bf16x8*)&lb[srow][shalf * 16]     = bp[0];
    *(bf16x8*)&lb[srow][shalf * 16 + 8] = bp[1];
    // warm next K-tile while this one is consumed (global_prefetch_b8)
    if (k0 + 32 < K) {
      __builtin_prefetch(arow + k0 + 32, 0, 1);
      __builtin_prefetch(brow + k0 + 32, 0, 1);
    }
    __syncthreads();

    v16bf afr[2], bfr[4];
#pragma unroll
    for (int tm = 0; tm < 2; ++tm)
      afr[tm] = load_frag(&la[wm * 32 + tm * 16 + l16][0], half);
#pragma unroll
    for (int tn = 0; tn < 4; ++tn)
      bfr[tn] = load_frag(&lb[wn * 64 + tn * 16 + l16][0], half);
#pragma unroll
    for (int tm = 0; tm < 2; ++tm)
#pragma unroll
      for (int tn = 0; tn < 4; ++tn)
        acc[tm][tn] = wmma_bf16(afr[tm], bfr[tn], acc[tm][tn]);
    __syncthreads();
  }

  // ---- epilogue: bias + store ----
#pragma unroll
  for (int tm = 0; tm < 2; ++tm) {
#pragma unroll
    for (int tn = 0; tn < 4; ++tn) {
      union { v8f v; float f[8]; } u; u.v = acc[tm][tn];
      const int col = n0 + wn * 64 + tn * 16 + l16;
      const float bv = bias[col];
#pragma unroll
      for (int e = 0; e < 8; ++e) {
        const int row = m0 + wm * 32 + tm * 16 + (half ? e + 8 : e);
        store_out(&C[(size_t)row * N + col], u.f[e] + bv);
      }
    }
  }
}

// ---------------------------------------------------------------------------
// Flash-style attention.  Block = 128 queries x 1 head; 8 waves, each wave
// owns a 16-query strip; keys streamed in 32-wide chunks with online softmax.
// K chunk staged via async global->LDS DMA when available.
// ---------------------------------------------------------------------------
__global__ __launch_bounds__(256)
void attention_kernel(const __bf16* __restrict__ Q, const __bf16* __restrict__ Kb,
                      const __bf16* __restrict__ Vb, float* __restrict__ O) {
  __shared__ __bf16 sk[32][64];        // K chunk, row-major [key][d]     (4KB)
  __shared__ __bf16 svt[64][48];       // V chunk transposed [d][key]     (6KB)
  __shared__ __bf16 sp[8][16][48];     // per-wave P tile [q][key]       (12KB)

  const int tid = threadIdx.x;
  const int wave = tid >> 5, lane = tid & 31;
  const int half = lane >> 4, l16 = lane & 15;
  const int q0 = blockIdx.x * 128;
  const int h  = blockIdx.y;
  const int b  = blockIdx.z;

  // Q fragments for this wave's 16-row strip, straight from global (aligned).
  const int qrow = q0 + wave * 16 + l16;
  const __bf16* qptr = Q + (size_t)(b * SEQ + qrow) * EMBED + h * DHEAD;
  const v16bf qa0 = load_frag(qptr, half);        // depth 0..31
  const v16bf qa1 = load_frag(qptr + 32, half);   // depth 32..63

  float m_i[8], l_i[8];
#pragma unroll
  for (int e = 0; e < 8; ++e) { m_i[e] = -3.0e38f; l_i[e] = 0.f; }
  v8f oacc[4];
#pragma unroll
  for (int t = 0; t < 4; ++t) oacc[t] = zero8();

  const int srK = tid >> 3, sc8 = tid & 7;        // staging: 32 rows x 8 chunks

  for (int kv0 = 0; kv0 < SEQ; kv0 += 32) {
    // ---- stage K rows (async DMA to LDS if available) and V transposed ----
    const size_t kvoff = (size_t)(b * SEQ + kv0 + srK) * EMBED + h * DHEAD + sc8 * 8;
#if HAVE_ASYNC_LDS
    __builtin_amdgcn_global_load_async_to_lds_b128(
        (GAS v4i*)(Kb + kvoff),
        (LAS v4i*)&sk[srK][sc8 * 8], 0, 0);
#else
    *(bf16x8*)&sk[srK][sc8 * 8] = *(const bf16x8*)(Kb + kvoff);
#endif
    bf16x8 vv = *(const bf16x8*)(Vb + kvoff);
#pragma unroll
    for (int j = 0; j < 8; ++j) svt[sc8 * 8 + j][srK] = vv[j];
    // warm next chunk (global_prefetch_b8)
    if (kv0 + 32 < SEQ) {
      __builtin_prefetch(Kb + kvoff + (size_t)32 * EMBED, 0, 1);
      __builtin_prefetch(Vb + kvoff + (size_t)32 * EMBED, 0, 1);
    }
#if HAVE_ASYNC_LDS
    // Each wave must retire its own async transfers before signaling.
#if __has_builtin(__builtin_amdgcn_s_wait_asynccnt)
    __builtin_amdgcn_s_wait_asynccnt(0);
#else
    asm volatile("s_wait_asynccnt 0" ::: "memory");
#endif
#endif
    __syncthreads();

    // ---- scores: 16q x 32k = two 16x16 C tiles, depth 64 in two WMMAs ----
    v8f s0 = zero8(), s1 = zero8();
    {
      v16bf kb00 = load_frag(&sk[l16][0], half);
      v16bf kb01 = load_frag(&sk[l16][32], half);
      s0 = wmma_bf16(qa0, kb00, s0);
      s0 = wmma_bf16(qa1, kb01, s0);
      v16bf kb10 = load_frag(&sk[16 + l16][0], half);
      v16bf kb11 = load_frag(&sk[16 + l16][32], half);
      s1 = wmma_bf16(qa0, kb10, s1);
      s1 = wmma_bf16(qa1, kb11, s1);
    }

    // ---- online softmax (rows live in VGPR index; cols across 16 lanes) ----
    union { v8f v; float f[8]; } u0, u1;
    u0.v = s0; u1.v = s1;
    const float scale = 0.125f;   // 1/sqrt(64)
    float alpha[8];
#pragma unroll
    for (int e = 0; e < 8; ++e) {
      float a = u0.f[e] * scale;
      float c = u1.f[e] * scale;
      float cm = fmaxf(a, c);
      cm = fmaxf(cm, __shfl_xor(cm, 1, 32));
      cm = fmaxf(cm, __shfl_xor(cm, 2, 32));
      cm = fmaxf(cm, __shfl_xor(cm, 4, 32));
      cm = fmaxf(cm, __shfl_xor(cm, 8, 32));
      const float mn = fmaxf(m_i[e], cm);
      alpha[e] = __expf(m_i[e] - mn);
      m_i[e] = mn;
      const float p0 = __expf(a - mn);
      const float p1 = __expf(c - mn);
      float cs = p0 + p1;
      cs += __shfl_xor(cs, 1, 32);
      cs += __shfl_xor(cs, 2, 32);
      cs += __shfl_xor(cs, 4, 32);
      cs += __shfl_xor(cs, 8, 32);
      l_i[e] = l_i[e] * alpha[e] + cs;
      const int row = half ? e + 8 : e;            // C-layout row for element e
      sp[wave][row][l16]      = f2bf(p0);
      sp[wave][row][16 + l16] = f2bf(p1);
    }

    // rescale running output by alpha (per-row)
#pragma unroll
    for (int t = 0; t < 4; ++t) {
      union { v8f v; float f[8]; } uo; uo.v = oacc[t];
#pragma unroll
      for (int e = 0; e < 8; ++e) uo.f[e] *= alpha[e];
      oacc[t] = uo.v;
    }

    // Make the per-wave P writes visible before re-reading in A-frag layout.
    asm volatile("s_wait_dscnt 0" ::: "memory");
    const v16bf pa = load_frag(&sp[wave][l16][0], half);   // 16q x 32k A-frag

    // ---- P @ V : depth 64 output = 4 N-tiles ----
#pragma unroll
    for (int t = 0; t < 4; ++t) {
      v16bf vbf = load_frag(&svt[t * 16 + l16][0], half);
      oacc[t] = wmma_bf16(pa, vbf, oacc[t]);
    }
    __syncthreads();
  }

  // ---- normalize and write f32 concat output [B,S,D] ----
#pragma unroll
  for (int t = 0; t < 4; ++t) {
    union { v8f v; float f[8]; } uo; uo.v = oacc[t];
#pragma unroll
    for (int e = 0; e < 8; ++e) {
      const int row = half ? e + 8 : e;
      const int sr = q0 + wave * 16 + row;
      O[(size_t)(b * SEQ + sr) * EMBED + h * DHEAD + t * 16 + l16] =
          uo.f[e] / l_i[e];
    }
  }
}

// ---------------------------------------------------------------------------
extern "C" void kernel_launch(void* const* d_in, const int* in_sizes, int n_in,
                              void* d_out, int out_size, void* d_ws, size_t ws_size,
                              hipStream_t stream) {
  (void)in_sizes; (void)n_in; (void)out_size; (void)ws_size;
  const float* q  = (const float*)d_in[0];
  const float* k  = (const float*)d_in[1];
  const float* v  = (const float*)d_in[2];
  const float* Wq = (const float*)d_in[3];
  const float* bq = (const float*)d_in[4];
  const float* Wk = (const float*)d_in[5];
  const float* bk = (const float*)d_in[6];
  const float* Wv = (const float*)d_in[7];
  const float* bv = (const float*)d_in[8];
  const float* Wo = (const float*)d_in[9];
  const float* bo = (const float*)d_in[10];
  float* out = (float*)d_out;

  char* ws = (char*)d_ws;
  const size_t MB = 1ull << 20;
  __bf16* Wqt = (__bf16*)(ws + 0 * MB);    // 2MB each
  __bf16* Wkt = (__bf16*)(ws + 2 * MB);
  __bf16* Wvt = (__bf16*)(ws + 4 * MB);
  __bf16* Wot = (__bf16*)(ws + 6 * MB);
  __bf16* Qb  = (__bf16*)(ws + 8 * MB);    // 8MB each (B*S*D bf16)
  __bf16* Kb  = (__bf16*)(ws + 16 * MB);
  __bf16* Vb  = (__bf16*)(ws + 24 * MB);
  float*  Cc  = (float*)(ws + 32 * MB);    // 16MB concat f32

  const int M = BATCH * SEQ;               // 4096

  dim3 tb(32, 8), tg(EMBED / 32, EMBED / 32);
  wt_convert_kernel<<<tg, tb, 0, stream>>>(Wq, Wqt);
  wt_convert_kernel<<<tg, tb, 0, stream>>>(Wk, Wkt);
  wt_convert_kernel<<<tg, tb, 0, stream>>>(Wv, Wvt);
  wt_convert_kernel<<<tg, tb, 0, stream>>>(Wo, Wot);

  dim3 gg(EMBED / 128, M / 128);           // (8, 32)
  gemm_bias_kernel<__bf16><<<gg, 256, 0, stream>>>(q, Wqt, bq, Qb, M, EMBED, EMBED);
  gemm_bias_kernel<__bf16><<<gg, 256, 0, stream>>>(k, Wkt, bk, Kb, M, EMBED, EMBED);
  gemm_bias_kernel<__bf16><<<gg, 256, 0, stream>>>(v, Wvt, bv, Vb, M, EMBED, EMBED);

  dim3 ag(SEQ / 128, NHEAD, BATCH);        // (16, 16, 2)
  attention_kernel<<<ag, 256, 0, stream>>>(Qb, Kb, Vb, Cc);

  gemm_bias_kernel<float><<<gg, 256, 0, stream>>>(Cc, Wot, bo, out, M, EMBED, EMBED);
}
